// HebbianLayer_1348619731089
// MI455X (gfx1250) — compile-verified, matrix operand order checked by reading, not run on hardware
//
#include <hip/hip_runtime.h>

typedef __attribute__((ext_vector_type(16))) _Float16 v16h;
typedef __attribute__((ext_vector_type(8)))  float    v8f;

// ---------------------------------------------------------------------------
// Fixed problem shapes (from reference setup_inputs)
// ---------------------------------------------------------------------------
#define BATCH 4
#define SEQ   4096
#define DMODEL 512
#define DINNER 1024
#define MROWS (BATCH * SEQ)   // 16384
#define CHUNK 64
#define NCHUNK (SEQ / CHUNK)  // 64

__device__ __forceinline__ int lane_id() { return threadIdx.x & 31; }

// ---------------------------------------------------------------------------
// CDNA5 async copy global->LDS (CDNA5 ISA):
// per-lane 16B transfer, tracked by ASYNCcnt. LDS address = low 32 bits of the
// generic shared-space pointer (flat->LDS mapping truncates, ISA 10.2).
// ---------------------------------------------------------------------------
__device__ __forceinline__ void async_load_b128(unsigned int lds_byte_addr,
                                                const void* gaddr) {
  asm volatile("global_load_async_to_lds_b128 %0, %1, off"
               :
               : "v"(lds_byte_addr), "v"((unsigned long long)(uintptr_t)gaddr)
               : "memory");
}

__device__ __forceinline__ void wait_asynccnt0() {
  asm volatile("s_wait_asynccnt 0" ::: "memory");
}

__device__ __forceinline__ unsigned int lds_addr_of(const void* p) {
  return (unsigned int)(uintptr_t)p;
}

// ---------------------------------------------------------------------------
// WMMA fragment helpers (CDNA5 wave32 layouts, CDNA5 ISA §7.12.2)
// A 16x32 f16: lanes 0-15 -> M=lane, K={0..7}∪{16..23}; lanes 16-31 -> M=lane-16,
// K={8..15}∪{24..31}.  B 32x16 mirrored with N on lanes.  C/D f32: VGPR r holds
// (M=r, N=lane) for lanes 0-15 and (M=r+8, N=lane-16) for lanes 16-31.
// ---------------------------------------------------------------------------
__device__ __forceinline__ v16h frag_a_rowmajor(const _Float16* p, int ld) {
  const int l  = lane_id();
  const int m  = l & 15;
  const int k0 = (l & 16) ? 8 : 0;
  const _Float16* r = p + (size_t)m * ld;
  v16h f;
#pragma unroll
  for (int i = 0; i < 8; ++i) {
    f[i]     = r[k0 + i];
    f[8 + i] = r[16 + k0 + i];
  }
  return f;
}

// B fragment from K-major (KxN row-major) source: B[k,n] = p[k*ld + n].
__device__ __forceinline__ v16h frag_b_kmajor(const _Float16* p, int ld) {
  const int l  = lane_id();
  const int n  = l & 15;
  const int k0 = (l & 16) ? 8 : 0;
  v16h f;
#pragma unroll
  for (int i = 0; i < 8; ++i) {
    f[i]     = p[(size_t)(k0 + i) * ld + n];
    f[8 + i] = p[(size_t)(16 + k0 + i) * ld + n];
  }
  return f;
}

// B^T fragment from row-major f32 source with on-the-fly f16 conversion
// (used to feed the f32 LDS state S into WMMA).
__device__ __forceinline__ v16h frag_bt_f32(const float* p, int ld) {
  const int l  = lane_id();
  const int m  = l & 15;
  const int k0 = (l & 16) ? 8 : 0;
  const float* r = p + (size_t)m * ld;
  v16h f;
#pragma unroll
  for (int i = 0; i < 8; ++i) {
    f[i]     = (_Float16)r[k0 + i];
    f[8 + i] = (_Float16)r[16 + k0 + i];
  }
  return f;
}

// A fragment for (dk .* v)^T : A[m=d_local, k=j] = dk[jbase+j] * v[j*ld + m]
// (dk table precomputed in LDS — no transcendentals in the hot loop)
__device__ __forceinline__ v16h frag_a_dkv(const _Float16* p, int ld,
                                           const float* dk, int jbase) {
  const int l  = lane_id();
  const int m  = l & 15;
  const int k0 = (l & 16) ? 8 : 0;
  v16h f;
#pragma unroll
  for (int i = 0; i < 8; ++i) {
    const int j1 = k0 + i, j2 = 16 + k0 + i;
    f[i]     = (_Float16)(dk[jbase + j1] * (float)p[(size_t)j1 * ld + m]);
    f[8 + i] = (_Float16)(dk[jbase + j2] * (float)p[(size_t)j2 * ld + m]);
  }
  return f;
}

__device__ __forceinline__ v8f load_acc_f32(const float* p, int ld) {
  const int l  = lane_id();
  const int n  = l & 15;
  const int mb = (l & 16) ? 8 : 0;
  v8f c;
#pragma unroll
  for (int r = 0; r < 8; ++r) c[r] = p[(size_t)(mb + r) * ld + n];
  return c;
}

__device__ __forceinline__ void store_acc_f32(float* p, int ld, v8f c) {
  const int l  = lane_id();
  const int n  = l & 15;
  const int mb = (l & 16) ? 8 : 0;
#pragma unroll
  for (int r = 0; r < 8; ++r) p[(size_t)(mb + r) * ld + n] = c[r];
}

__device__ __forceinline__ void store_acc_f16(_Float16* p, int ld, v8f c) {
  const int l  = lane_id();
  const int n  = l & 15;
  const int mb = (l & 16) ? 8 : 0;
#pragma unroll
  for (int r = 0; r < 8; ++r) p[(size_t)(mb + r) * ld + n] = (_Float16)c[r];
}

__device__ __forceinline__ v8f wmma_f16(v16h a, v16h b, v8f c) {
  return __builtin_amdgcn_wmma_f32_16x16x32_f16(false, a, false, b,
                                                (short)0, c, false, false);
}

__device__ __forceinline__ float silu(float x) {
  return x / (1.0f + __expf(-x));
}

// ---------------------------------------------------------------------------
// Kernel 1: RMSNorm -> f16   (wave-per-row, wave32 shuffle reduction)
// ---------------------------------------------------------------------------
__global__ __launch_bounds__(256)
void rmsnorm_f16_kernel(const float* __restrict__ x,
                        const float* __restrict__ norm_w,
                        _Float16* __restrict__ xh) {
  const int row  = blockIdx.x * 8 + (threadIdx.x >> 5);
  const int lane = lane_id();
  const float* xr = x + (size_t)row * DMODEL;
  float v[16];
  float ss = 0.0f;
#pragma unroll
  for (int i = 0; i < 16; ++i) {
    v[i] = xr[lane + 32 * i];
    ss += v[i] * v[i];
  }
#pragma unroll
  for (int off = 16; off > 0; off >>= 1) ss += __shfl_xor(ss, off, 32);
  const float r = rsqrtf(ss * (1.0f / DMODEL) + 1e-5f);
  _Float16* o = xh + (size_t)row * DMODEL;
#pragma unroll
  for (int i = 0; i < 16; ++i)
    o[lane + 32 * i] = (_Float16)(v[i] * r * norm_w[lane + 32 * i]);
}

// ---------------------------------------------------------------------------
// Kernel 2: f32 -> f16 weight conversion
// ---------------------------------------------------------------------------
__global__ void f32_to_f16_kernel(const float* __restrict__ src,
                                  _Float16* __restrict__ dst, int n) {
  const int i = blockIdx.x * 256 + threadIdx.x;
  if (i < n) dst[i] = (_Float16)src[i];
}

// ---------------------------------------------------------------------------
// Kernel 3: WMMA GEMM  C[M,N] = A[M,K] @ Bt[N,K]^T   (block tile 128x64,
// 8 waves, wave-per-16-rows, 4 col accumulators, K-step 32)
// EPI: 1 = store f16, 2 = store f32 with residual add
// ---------------------------------------------------------------------------
template <int EPI>
__global__ __launch_bounds__(256)
void gemm_wmma_kernel(const _Float16* __restrict__ A,
                      const _Float16* __restrict__ Bt,
                      float* __restrict__ Cf,
                      _Float16* __restrict__ Ch,
                      const float* __restrict__ resid,
                      int M, int N, int Kdim) {
  const int wave = threadIdx.x >> 5;
  const int row0 = blockIdx.x * 128 + wave * 16;
  const int col0 = blockIdx.y * 64;
  v8f acc[4] = {v8f{}, v8f{}, v8f{}, v8f{}};
  const _Float16* arow = A + (size_t)row0 * Kdim;
  for (int ks = 0; ks < Kdim; ks += 32) {
    if (ks + 32 < Kdim)
      __builtin_prefetch(arow + ks + 32, 0, 3);   // global_prefetch_b8
    v16h a = frag_a_rowmajor(arow + ks, Kdim);
#pragma unroll
    for (int t = 0; t < 4; ++t) {
      v16h b = frag_a_rowmajor(Bt + (size_t)(col0 + t * 16) * Kdim + ks, Kdim);
      acc[t] = wmma_f16(a, b, acc[t]);
    }
  }
  const int l = lane_id(), n = l & 15, mb = (l & 16) ? 8 : 0;
#pragma unroll
  for (int t = 0; t < 4; ++t) {
    if (EPI == 1) {
      store_acc_f16(Ch + (size_t)row0 * N + col0 + t * 16, N, acc[t]);
    } else {
      float* co = Cf + (size_t)row0 * N + col0 + t * 16;
      const float* rr = resid + (size_t)row0 * N + col0 + t * 16;
#pragma unroll
      for (int r = 0; r < 8; ++r)
        co[(size_t)(mb + r) * N + n] = acc[t][r] + rr[(size_t)(mb + r) * N + n];
    }
  }
}

// ---------------------------------------------------------------------------
// Kernel 4: causal depthwise conv1d (K=4) + dual SiLU gate -> f16 (in place ok)
// ---------------------------------------------------------------------------
__global__ __launch_bounds__(256)
void conv_gate_kernel(const _Float16* __restrict__ val_h,
                      const _Float16* __restrict__ gate_h,
                      const float* __restrict__ conv_w,
                      const float* __restrict__ conv_b,
                      _Float16* __restrict__ h_h) {
  const size_t idx = (size_t)blockIdx.x * 256 + threadIdx.x; // over M*DINNER
  const int di = (int)(idx & (DINNER - 1));
  const size_t bl = idx >> 10;          // row in [0, MROWS)
  const int l = (int)(bl & (SEQ - 1));  // position within sequence
  float acc = conv_b[di];
#pragma unroll
  for (int i = 0; i < 4; ++i) {
    const int ls = l - 3 + i;
    if (ls >= 0)
      acc += (float)val_h[(bl - (size_t)(3 - i)) * DINNER + di] * conv_w[di * 4 + i];
  }
  const float g = (float)gate_h[idx];
  h_h[idx] = (_Float16)(silu(acc) * silu(g));
}

// ---------------------------------------------------------------------------
// Stage one chunk's k-panel (64KB, contiguous) and v-tile (8KB, 128B rows)
// into LDS via CDNA5 async DMA (global_load_async_to_lds_b128, ASYNCcnt).
// ---------------------------------------------------------------------------
__device__ __forceinline__ void stage_chunk(const _Float16* __restrict__ kg,
                                            const _Float16* __restrict__ vg,
                                            _Float16* kdst, _Float16* vdst,
                                            int tid) {
  const unsigned int klds = lds_addr_of(kdst);
  const char* ksrc = (const char*)kg;
#pragma unroll
  for (int it = 0; it < 16; ++it) {
    const unsigned int off = (unsigned int)((it * 256 + tid) * 16);
    async_load_b128(klds + off, ksrc + off);
  }
  const unsigned int vlds = lds_addr_of(vdst);
#pragma unroll
  for (int it = 0; it < 2; ++it) {
    const int e8 = (it * 256 + tid) * 8;        // f16 element index in 64x64 tile
    const int row = e8 >> 6, col = e8 & 63;
    async_load_b128(vlds + (unsigned int)(row * 64 + col) * 2,
                    vg + (size_t)row * DMODEL + col);
  }
}

// ---------------------------------------------------------------------------
// Kernel 5: chunked Hebbian fast-weight scan.
// One block per (batch, 64-row d-tile). LDS (~304 KB of the 320 KB WGP pool):
//   S[64][512] f32 state, double-buffered k-panel + v-tile (async-DMA'd),
//   score tiles, decay tables. Per chunk: scores=qk^T (WMMA), decay-mask,
//   intra=scores*v (WMMA), inter=q*S^T (WMMA, dq-scaled),
//   S = gamma^C*S + (dk.*v)^T k (WMMA accumulating into f32 LDS state).
// Next chunk's DMA is issued before compute so ASYNC transfers overlap WMMA.
// ---------------------------------------------------------------------------
__global__ __launch_bounds__(256)
void hebbian_attn_kernel(const _Float16* __restrict__ qh,
                         const _Float16* __restrict__ kh,
                         const _Float16* __restrict__ vh,
                         const float* __restrict__ alpha_logit,
                         _Float16* __restrict__ yh) {
  __shared__ float    S[CHUNK * DMODEL];        // 128 KB state tile
  __shared__ _Float16 kb0[CHUNK * DMODEL];      // 64 KB k-panel (buf 0)
  __shared__ _Float16 kb1[CHUNK * DMODEL];      // 64 KB k-panel (buf 1)
  __shared__ _Float16 vb0[CHUNK * CHUNK];       // 8 KB v-tile (buf 0)
  __shared__ _Float16 vb1[CHUNK * CHUNK];       // 8 KB v-tile (buf 1)
  __shared__ float    scf[CHUNK * CHUNK];       // 16 KB raw scores
  __shared__ _Float16 sch[CHUNK * CHUNK];       // 8 KB masked scores (f16)
  __shared__ float    gpow[CHUNK];              // gamma^delta
  __shared__ float    dkb[CHUNK];               // gamma^(63-j)
  __shared__ float    dqb[CHUNK];               // gamma^(c+1)

  const int b    = blockIdx.x >> 3;
  const int d0   = (blockIdx.x & 7) * 64;
  const int tid  = threadIdx.x;
  const int wave = tid >> 5;

  const float al    = alpha_logit[0];
  const float gamma = 1.0f - 1.0f / (1.0f + __expf(-al));
  const float gC    = __powf(gamma, (float)CHUNK);

  if (tid < CHUNK) {
    gpow[tid] = __powf(gamma, (float)tid);
    dkb[tid]  = __powf(gamma, (float)(CHUNK - 1 - tid));
    dqb[tid]  = __powf(gamma, (float)(tid + 1));
  }
  for (int i = tid; i < CHUNK * DMODEL; i += 256) S[i] = 0.0f;

  const size_t bbase = (size_t)b * SEQ;
  stage_chunk(kh + bbase * DMODEL, vh + bbase * DMODEL + d0, kb0, vb0, tid);
  wait_asynccnt0();
  __syncthreads();

  int cur = 0;
  for (int ch = 0; ch < NCHUNK; ++ch) {
    const size_t m0 = bbase + (size_t)ch * CHUNK;
    const _Float16* qb = qh + m0 * DMODEL;
    const _Float16* kcur = cur ? kb1 : kb0;
    const _Float16* vcur = cur ? vb1 : vb0;

    // kick off next chunk's async DMA into the other buffer (overlaps WMMA)
    if (ch + 1 < NCHUNK) {
      const size_t m1 = m0 + CHUNK;
      stage_chunk(kh + m1 * DMODEL, vh + m1 * DMODEL + d0,
                  cur ? kb0 : kb1, cur ? vb0 : vb1, tid);
    }

    // ---- 1) scores = q k^T  (16 tiles of 16x16, 2 per wave, K=512)
#pragma unroll
    for (int t2 = 0; t2 < 2; ++t2) {
      const int t = wave * 2 + t2, ti = t >> 2, tj = t & 3;
      v8f acc = {};
      for (int ks = 0; ks < DMODEL; ks += 32) {
        v16h a  = frag_a_rowmajor(qb + (size_t)ti * 16 * DMODEL + ks, DMODEL);
        v16h bf = frag_a_rowmajor(kcur + (size_t)tj * 16 * DMODEL + ks, DMODEL);
        acc = wmma_f16(a, bf, acc);
      }
      store_acc_f32(scf + ti * 16 * CHUNK + tj * 16, CHUNK, acc);
    }
    __syncthreads();

    // ---- 2) causal decay mask + f16 convert (table lookup, no pow)
    for (int i = tid; i < CHUNK * CHUNK; i += 256) {
      const int r = i >> 6, c = i & 63;
      const float vv = (c <= r) ? scf[i] * gpow[r - c] : 0.0f;
      sch[i] = (_Float16)vv;
    }
    __syncthreads();

    // ---- 3) y = intra + dq * inter   (16 tiles, 2 per wave)
#pragma unroll
    for (int t2 = 0; t2 < 2; ++t2) {
      const int t = wave * 2 + t2, ti = t >> 2, tj = t & 3;
      v8f intra = {};
#pragma unroll
      for (int ks = 0; ks < CHUNK; ks += 32) {
        v16h a  = frag_a_rowmajor(sch + ti * 16 * CHUNK + ks, CHUNK);
        v16h bf = frag_b_kmajor(vcur + (size_t)ks * CHUNK + tj * 16, CHUNK);
        intra = wmma_f16(a, bf, intra);
      }
      v8f inter = {};
      for (int ks = 0; ks < DMODEL; ks += 32) {
        v16h a  = frag_a_rowmajor(qb + (size_t)ti * 16 * DMODEL + ks, DMODEL);
        v16h bf = frag_bt_f32(S + (size_t)tj * 16 * DMODEL + ks, DMODEL);
        inter = wmma_f16(a, bf, inter);
      }
      const int l = lane_id(), n = l & 15, mb = (l & 16) ? 8 : 0;
      _Float16* yo = yh + (m0 + (size_t)ti * 16) * DMODEL + d0 + tj * 16;
#pragma unroll
      for (int r = 0; r < 8; ++r) {
        const int cpos = ti * 16 + mb + r;
        yo[(size_t)(mb + r) * DMODEL + n] =
            (_Float16)(intra[r] + dqb[cpos] * inter[r]);
      }
    }
    __syncthreads();  // all reads of S complete before update

    // ---- 4) S = gC * S + (dk .* v)^T k   (128 tiles, 16 per wave)
    for (int t2 = 0; t2 < 16; ++t2) {
      const int t = wave * 16 + t2, td = t >> 5, te = t & 31;
      float* Sp = S + (size_t)td * 16 * DMODEL + te * 16;
      v8f c = load_acc_f32(Sp, DMODEL);
#pragma unroll
      for (int r = 0; r < 8; ++r) c[r] *= gC;
#pragma unroll
      for (int ks = 0; ks < CHUNK; ks += 32) {
        v16h a  = frag_a_dkv(vcur + (size_t)ks * CHUNK + td * 16, CHUNK,
                             dkb, ks);
        v16h bf = frag_b_kmajor(kcur + (size_t)ks * DMODEL + te * 16, DMODEL);
        c = wmma_f16(a, bf, c);
      }
      store_acc_f32(Sp, DMODEL, c);
    }
    wait_asynccnt0();   // next chunk's DMA complete before buffer swap
    __syncthreads();
    cur ^= 1;
  }
}

// ---------------------------------------------------------------------------
// Host-side launch. Workspace layout (reuse; peak ~104 MB):
//   [0,16)MB   xn_h  -> later y_h
//   [16,48)MB  val_h -> later q_h(16..32), k_h(32..48)
//   [48,80)MB  gate_h -> h_h (in place) -> v_h(48..64)
//   [80,96)MB  u_h
//   [96,104)MB f16 weights
// ---------------------------------------------------------------------------
extern "C" void kernel_launch(void* const* d_in, const int* in_sizes, int n_in,
                              void* d_out, int out_size, void* d_ws, size_t ws_size,
                              hipStream_t stream) {
  const float* x      = (const float*)d_in[0];
  const float* norm_w = (const float*)d_in[1];
  const float* proj_w = (const float*)d_in[2];
  const float* gate_w = (const float*)d_in[3];
  const float* conv_w = (const float*)d_in[4];
  const float* conv_b = (const float*)d_in[5];
  const float* out_w  = (const float*)d_in[6];
  const float* wq     = (const float*)d_in[7];
  const float* wk     = (const float*)d_in[8];
  const float* wv     = (const float*)d_in[9];
  const float* wo     = (const float*)d_in[10];
  const float* alpha  = (const float*)d_in[11];
  float* out = (float*)d_out;

  char* ws = (char*)d_ws;
  const size_t MB = 1ull << 20;
  _Float16* xn_h   = (_Float16*)(ws + 0);
  _Float16* y_h    = (_Float16*)(ws + 0);          // reuse (xn dead)
  _Float16* val_h  = (_Float16*)(ws + 16 * MB);
  _Float16* q_h    = (_Float16*)(ws + 16 * MB);    // reuse (val dead)
  _Float16* k_h    = (_Float16*)(ws + 32 * MB);
  _Float16* gate_h = (_Float16*)(ws + 48 * MB);    // h written in place
  _Float16* v_h    = (_Float16*)(ws + 48 * MB);    // reuse (h dead)
  _Float16* u_h    = (_Float16*)(ws + 80 * MB);
  _Float16* pwh    = (_Float16*)(ws + 96 * MB);
  _Float16* gwh    = pwh + (size_t)DINNER * DMODEL;
  _Float16* owh    = gwh + (size_t)DINNER * DMODEL;
  _Float16* wqh    = owh + (size_t)DMODEL * DINNER;
  _Float16* wkh    = wqh + (size_t)DMODEL * DMODEL;
  _Float16* wvh    = wkh + (size_t)DMODEL * DMODEL;
  _Float16* woh    = wvh + (size_t)DMODEL * DMODEL;

  const dim3 blk(256);

  // weight conversion f32 -> f16
  const int nw1 = DINNER * DMODEL, nw2 = DMODEL * DMODEL;
  f32_to_f16_kernel<<<(nw1 + 255) / 256, blk, 0, stream>>>(proj_w, pwh, nw1);
  f32_to_f16_kernel<<<(nw1 + 255) / 256, blk, 0, stream>>>(gate_w, gwh, nw1);
  f32_to_f16_kernel<<<(nw1 + 255) / 256, blk, 0, stream>>>(out_w,  owh, nw1);
  f32_to_f16_kernel<<<(nw2 + 255) / 256, blk, 0, stream>>>(wq, wqh, nw2);
  f32_to_f16_kernel<<<(nw2 + 255) / 256, blk, 0, stream>>>(wk, wkh, nw2);
  f32_to_f16_kernel<<<(nw2 + 255) / 256, blk, 0, stream>>>(wv, wvh, nw2);
  f32_to_f16_kernel<<<(nw2 + 255) / 256, blk, 0, stream>>>(wo, woh, nw2);

  // 1) RMSNorm
  rmsnorm_f16_kernel<<<MROWS / 8, blk, 0, stream>>>(x, norm_w, xn_h);

  // 2) val / gate projections [16384,1024]
  gemm_wmma_kernel<1><<<dim3(MROWS / 128, DINNER / 64), blk, 0, stream>>>(
      xn_h, pwh, nullptr, val_h, nullptr, MROWS, DINNER, DMODEL);
  gemm_wmma_kernel<1><<<dim3(MROWS / 128, DINNER / 64), blk, 0, stream>>>(
      xn_h, gwh, nullptr, gate_h, nullptr, MROWS, DINNER, DMODEL);

  // 3) causal conv + SiLU gating (h written in place over gate buffer)
  conv_gate_kernel<<<(MROWS * DINNER) / 256, blk, 0, stream>>>(
      val_h, gate_h, conv_w, conv_b, gate_h);

  // 4) u = h @ out_w^T  [16384,512]
  gemm_wmma_kernel<1><<<dim3(MROWS / 128, DMODEL / 64), blk, 0, stream>>>(
      gate_h, owh, nullptr, u_h, nullptr, MROWS, DMODEL, DINNER);

  // 5) q / k / v  [16384,512]
  gemm_wmma_kernel<1><<<dim3(MROWS / 128, DMODEL / 64), blk, 0, stream>>>(
      u_h, wqh, nullptr, q_h, nullptr, MROWS, DMODEL, DMODEL);
  gemm_wmma_kernel<1><<<dim3(MROWS / 128, DMODEL / 64), blk, 0, stream>>>(
      u_h, wkh, nullptr, k_h, nullptr, MROWS, DMODEL, DMODEL);
  gemm_wmma_kernel<1><<<dim3(MROWS / 128, DMODEL / 64), blk, 0, stream>>>(
      u_h, wvh, nullptr, v_h, nullptr, MROWS, DMODEL, DMODEL);

  // 6) Hebbian chunked scan: 4 batches x 8 d-tiles = 32 blocks
  hebbian_attn_kernel<<<BATCH * 8, blk, 0, stream>>>(q_h, k_h, v_h, alpha, y_h);

  // 7) out = x + y @ wo^T
  gemm_wmma_kernel<2><<<dim3(MROWS / 128, DMODEL / 64), blk, 0, stream>>>(
      y_h, woh, out, nullptr, x, MROWS, DMODEL, DMODEL);
}
